// SelfAttention_48687749267735
// MI455X (gfx1250) — matrix operand order repaired and checked by reading, not compiled, not run on hardware
//
#include <hip/hip_runtime.h>
#include <hip/hip_bf16.h>

// ---------------------------------------------------------------------------
// Self-attention forward for MI455X (gfx1250, wave32, WMMA).
//   1) fp32 -> bf16 conversions (weights are [out,in]: K-contiguous, so WMMA
//      A/B fragments are two 16B row loads per lane, no transposes)
//   2) GEMM  qkv[4096,3072] = feat . qkv_w^T + qkv_b    (bf16 out)
//      wave tile 64x32 (4x2 WMMA tiles), workgroup 128x128
//   3) flash-attention per (b,h): 32-key blocks, K tile staged with
//      GLOBAL_LOAD_ASYNC_TO_LDS_B128 (ASYNCcnt), scores + PV on
//      v_wmma_f32_16x16x32_bf16, online softmax via shfl_xor
//   4) GEMM  out[4096,1024] = attn . out_w^T + out_b    (fp32 out)
// ---------------------------------------------------------------------------

typedef __attribute__((ext_vector_type(16))) __bf16 v16bf;
typedef __attribute__((ext_vector_type(8)))  float  v8f;

#define BATCH   2
#define S_LEN   2048
#define DMODEL  1024
#define NHEADS  16
#define HDIM    64
#define QKV_N   (3 * DMODEL)

__device__ __forceinline__ unsigned short f2bf(float f) {
  unsigned int u = __float_as_uint(f);
  u += 0x7fffu + ((u >> 16) & 1u);          // round-to-nearest-even
  return (unsigned short)(u >> 16);
}

union Frag {
  v16bf v;
  uint4 q[2];
  unsigned short us[16];
};

// ---------------------------------------------------------------------------
// fp32 -> bf16 elementwise conversion
// ---------------------------------------------------------------------------
__global__ void cvt_f32_bf16(const float* __restrict__ in,
                             unsigned short* __restrict__ out, int n) {
  int i = blockIdx.x * blockDim.x + threadIdx.x;
  int stride = gridDim.x * blockDim.x;
  for (; i < n; i += stride) out[i] = f2bf(in[i]);
}

// ---------------------------------------------------------------------------
// GEMM: C[M,N] = A[M,K] . B[N,K]^T + bias[N]
// A,B bf16 row-major (K contiguous). 256 threads = 8 waves arranged 2(M)x4(N);
// wave tile 64x32 (4x2 WMMA tiles) -> workgroup tile 128x128.
// Requires M%128==0, N%128==0, K%32==0.
// ---------------------------------------------------------------------------
__global__ __launch_bounds__(256)
void gemm_bf16(const unsigned short* __restrict__ A,
               const unsigned short* __restrict__ B,
               const float* __restrict__ bias,
               void* __restrict__ Cout,
               int M, int N, int K, int c_bf16) {
  const int lane = threadIdx.x & 31;
  const int wave = threadIdx.x >> 5;
  const int wm = wave >> 2, wn = wave & 3;          // 2 x 4 wave grid
  const int m0 = blockIdx.x * 128 + wm * 64;        // 4 M-tiles of 16
  const int n0 = blockIdx.y * 128 + wn * 32;        // 2 N-tiles of 16
  const int g = lane >> 4, l15 = lane & 15;

  const v8f vzero = {0.f, 0.f, 0.f, 0.f, 0.f, 0.f, 0.f, 0.f};
  v8f acc[4][2];
#pragma unroll
  for (int i = 0; i < 4; ++i)
#pragma unroll
    for (int j = 0; j < 2; ++j) acc[i][j] = vzero;

  const unsigned short* aRow[4];
  const unsigned short* bRow[2];
#pragma unroll
  for (int i = 0; i < 4; ++i) aRow[i] = A + (size_t)(m0 + i * 16 + l15) * K;
#pragma unroll
  for (int j = 0; j < 2; ++j) bRow[j] = B + (size_t)(n0 + j * 16 + l15) * K;

  for (int k0 = 0; k0 < K; k0 += 32) {
    Frag a[4], b[2];
#pragma unroll
    for (int i = 0; i < 4; ++i) {
      // A fragment (16x32): lane row m=l15, K chunks {g*8..+7, 16+g*8..+7}
      a[i].q[0] = *(const uint4*)(aRow[i] + k0 + g * 8);
      a[i].q[1] = *(const uint4*)(aRow[i] + k0 + 16 + g * 8);
    }
#pragma unroll
    for (int j = 0; j < 2; ++j) {
      // B fragment (32x16): lane col n=l15, K chunks {g*8..+7, 16+g*8..+7}
      b[j].q[0] = *(const uint4*)(bRow[j] + k0 + g * 8);
      b[j].q[1] = *(const uint4*)(bRow[j] + k0 + 16 + g * 8);
    }
#pragma unroll
    for (int i = 0; i < 4; ++i)
#pragma unroll
      for (int j = 0; j < 2; ++j)
        acc[i][j] = __builtin_amdgcn_wmma_f32_16x16x32_bf16(
            false, a[i].v, false, b[j].v, (short)0, acc[i][j], false, false);
  }

  // C layout: lane holds column n=l15 of its 16-wide tile; rows r + 8*g.
#pragma unroll
  for (int j = 0; j < 2; ++j) {
    const int col = n0 + j * 16 + l15;
    const float bn = bias[col];
#pragma unroll
    for (int i = 0; i < 4; ++i) {
      const int rowBase = m0 + i * 16 + 8 * g;
#pragma unroll
      for (int r = 0; r < 8; ++r) {
        const float val = acc[i][j][r] + bn;
        const size_t off = (size_t)(rowBase + r) * N + col;
        if (c_bf16)
          ((unsigned short*)Cout)[off] = f2bf(val);
        else
          ((float*)Cout)[off] = val;
      }
    }
  }
}

// ---------------------------------------------------------------------------
// Flash attention: grid (S_LEN/64, BATCH*NHEADS), block 128 (4 waves).
// Each wave owns 16 queries; keys/values staged in 32-key blocks through LDS.
// K tile staged via GLOBAL_LOAD_ASYNC_TO_LDS_B128 (no VGPR round-trip).
// qkv is bf16 [BATCH*S_LEN, 3*DMODEL]; out is bf16 [BATCH*S_LEN, DMODEL].
// ---------------------------------------------------------------------------
__global__ __launch_bounds__(128)
void attn_flash(const unsigned short* __restrict__ qkv,
                unsigned short* __restrict__ outB,
                const int* __restrict__ causal_flag) {
  __shared__ unsigned short Kt[32][64];       // [key][e]   row-major
  __shared__ unsigned short Vt[64][32];       // [e][key]   transposed
  __shared__ unsigned short Pw[4][16][32];    // per-wave P scratch [q][key]

  const int causal = *causal_flag;
  const int tid = threadIdx.x;
  const int lane = tid & 31, wave = tid >> 5;
  const int g = lane >> 4, l15 = lane & 15;
  const int bh = blockIdx.y;
  const int bi = bh / NHEADS, h = bh % NHEADS;
  const int qbase = blockIdx.x * 64 + wave * 16;

  const size_t rowStride = QKV_N;
  const unsigned short* qPtr =
      qkv + (size_t)(bi * S_LEN + qbase + l15) * rowStride + h * HDIM;

  // Q A-fragments for head-dim chunks ek=0,1 (each 16x32)
  Frag qf[2];
#pragma unroll
  for (int ek = 0; ek < 2; ++ek) {
    qf[ek].q[0] = *(const uint4*)(qPtr + ek * 32 + g * 8);
    qf[ek].q[1] = *(const uint4*)(qPtr + ek * 32 + 16 + g * 8);
  }

  const v8f vzero = {0.f, 0.f, 0.f, 0.f, 0.f, 0.f, 0.f, 0.f};
  v8f acc_o[4];
#pragma unroll
  for (int et = 0; et < 4; ++et) acc_o[et] = vzero;
  float mrow[8], lrow[8];
#pragma unroll
  for (int r = 0; r < 8; ++r) { mrow[r] = -1e30f; lrow[r] = 0.f; }

  const int trow = tid >> 2;           // 0..31 : key row staged by this thread
  const int tcol = (tid & 3) * 16;     // 0,16,32,48 : e-offset
  const unsigned short* kBase =
      qkv + (size_t)(bi * S_LEN) * rowStride + DMODEL + h * HDIM;
  const unsigned short* vBase =
      qkv + (size_t)(bi * S_LEN) * rowStride + 2 * DMODEL + h * HDIM;
  // LDS byte address for async-load VDST (aperture maps addr[31:0] -> LDS)
  const unsigned ldsK = (unsigned)(size_t)&Kt[trow][tcol];
  const float scale = 0.125f;          // 1/sqrt(HDIM)

  for (int kb = 0; kb < S_LEN; kb += 32) {
    // ---- stage K tile directly into LDS via async copy (ASYNCcnt) ----
    const unsigned short* kp = kBase + (size_t)(kb + trow) * rowStride + tcol;
    asm volatile(
        "global_load_async_to_lds_b128 %0, %2, off\n\t"
        "global_load_async_to_lds_b128 %1, %3, off"
        :
        : "v"(ldsK), "v"(ldsK + 16),
          "v"((unsigned long long)(size_t)kp),
          "v"((unsigned long long)(size_t)(kp + 8))
        : "memory");

    // ---- stage V tile transposed (VGPR path; transpose not expressible
    //      as a straight async copy) ----
    const unsigned short* vp = vBase + (size_t)(kb + trow) * rowStride + tcol;
    Frag vtmp;
    vtmp.q[0] = *(const uint4*)(vp);
    vtmp.q[1] = *(const uint4*)(vp + 8);
#pragma unroll
    for (int j = 0; j < 16; ++j) Vt[tcol + j][trow] = vtmp.us[j];

    // async ops are invisible to the compiler's barrier lowering
    asm volatile("s_wait_asynccnt 0x0" ::: "memory");
    __syncthreads();

    // ---- S = Q . K^T : two 16x16 score tiles (keys kb+0..15, kb+16..31) ----
    v8f sfr[2];
#pragma unroll
    for (int nt = 0; nt < 2; ++nt) {
      sfr[nt] = vzero;
#pragma unroll
      for (int ek = 0; ek < 2; ++ek) {
        Frag kf;   // B fragment: lane col = key (nt*16+l15), K chunk over e
        kf.q[0] = *(const uint4*)(&Kt[nt * 16 + l15][ek * 32 + g * 8]);
        kf.q[1] = *(const uint4*)(&Kt[nt * 16 + l15][ek * 32 + 16 + g * 8]);
        sfr[nt] = __builtin_amdgcn_wmma_f32_16x16x32_bf16(
            false, qf[ek].v, false, kf.v, (short)0, sfr[nt], false, false);
      }
    }

    // ---- online softmax (rows r+8g live on this lane's 16-lane half) ----
#pragma unroll
    for (int r = 0; r < 8; ++r) {
      float s0 = sfr[0][r] * scale;
      float s1 = sfr[1][r] * scale;
      if (causal) {
        const int qrow = qbase + r + 8 * g;
        if (kb + l15 > qrow)      s0 = -1e30f;
        if (kb + 16 + l15 > qrow) s1 = -1e30f;
      }
      float mb = fmaxf(s0, s1);
      mb = fmaxf(mb, __shfl_xor(mb, 1, 32));
      mb = fmaxf(mb, __shfl_xor(mb, 2, 32));
      mb = fmaxf(mb, __shfl_xor(mb, 4, 32));
      mb = fmaxf(mb, __shfl_xor(mb, 8, 32));
      const float mnew = fmaxf(mrow[r], mb);
      const float alpha = __expf(mrow[r] - mnew);
      const float e0 = __expf(s0 - mnew);
      const float e1 = __expf(s1 - mnew);
      float psum = e0 + e1;
      psum += __shfl_xor(psum, 1, 32);
      psum += __shfl_xor(psum, 2, 32);
      psum += __shfl_xor(psum, 4, 32);
      psum += __shfl_xor(psum, 8, 32);
      lrow[r] = lrow[r] * alpha + psum;
      mrow[r] = mnew;
#pragma unroll
      for (int et = 0; et < 4; ++et) acc_o[et][r] *= alpha;
      // park P in per-wave LDS scratch to re-shape C-layout -> A-layout
      Pw[wave][r + 8 * g][l15]      = f2bf(e0);
      Pw[wave][r + 8 * g][16 + l15] = f2bf(e1);
    }
    // cross-lane LDS RAW hazard: wait for all lanes' ds stores
    asm volatile("s_wait_dscnt 0x0" ::: "memory");

    // ---- O += P . V ----
    Frag pf;   // A fragment 16x32: lane row m=l15
    pf.q[0] = *(const uint4*)(&Pw[wave][l15][g * 8]);
    pf.q[1] = *(const uint4*)(&Pw[wave][l15][16 + g * 8]);
#pragma unroll
    for (int et = 0; et < 4; ++et) {
      Frag vf;   // B fragment: lane col e = et*16+l15, keys {g*8..+7, 16+g*8..+7}
      vf.q[0] = *(const uint4*)(&Vt[et * 16 + l15][g * 8]);
      vf.q[1] = *(const uint4*)(&Vt[et * 16 + l15][16 + g * 8]);
      acc_o[et] = __builtin_amdgcn_wmma_f32_16x16x32_bf16(
          false, pf.v, false, vf.v, (short)0, acc_o[et], false, false);
    }
    __syncthreads();   // protect Kt/Vt before next block's staging
  }

  // ---- normalize and write bf16 [B*S, DMODEL] ----
#pragma unroll
  for (int r = 0; r < 8; ++r) {
    const float inv = 1.0f / lrow[r];
    const int qrow = qbase + r + 8 * g;
    const size_t base = (size_t)(bi * S_LEN + qrow) * DMODEL + h * HDIM;
#pragma unroll
    for (int et = 0; et < 4; ++et)
      outB[base + et * 16 + l15] = f2bf(acc_o[et][r] * inv);
  }
}

// ---------------------------------------------------------------------------
extern "C" void kernel_launch(void* const* d_in, const int* in_sizes, int n_in,
                              void* d_out, int out_size, void* d_ws,
                              size_t ws_size, hipStream_t stream) {
  const float* features = (const float*)d_in[0];
  const float* qkv_w    = (const float*)d_in[1];
  const float* qkv_b    = (const float*)d_in[2];
  const float* out_w    = (const float*)d_in[3];
  const float* out_b    = (const float*)d_in[4];
  const int*   causal   = (const int*)d_in[5];

  char* ws = (char*)d_ws;
  size_t off = 0;
  auto alloc = [&](size_t bytes) -> void* {
    void* p = ws + off;
    off = (off + bytes + 255) & ~(size_t)255;
    return p;
  };
  unsigned short* featB = (unsigned short*)alloc((size_t)BATCH * S_LEN * DMODEL * 2);
  unsigned short* qkvwB = (unsigned short*)alloc((size_t)QKV_N * DMODEL * 2);
  unsigned short* outwB = (unsigned short*)alloc((size_t)DMODEL * DMODEL * 2);
  unsigned short* qkvB  = (unsigned short*)alloc((size_t)BATCH * S_LEN * QKV_N * 2);
  unsigned short* attnB = (unsigned short*)alloc((size_t)BATCH * S_LEN * DMODEL * 2);

  cvt_f32_bf16<<<1024, 256, 0, stream>>>(features, featB, BATCH * S_LEN * DMODEL);
  cvt_f32_bf16<<<1024, 256, 0, stream>>>(qkv_w, qkvwB, QKV_N * DMODEL);
  cvt_f32_bf16<<<512, 256, 0, stream>>>(out_w, outwB, DMODEL * DMODEL);

  dim3 g1(BATCH * S_LEN / 128, QKV_N / 128);
  gemm_bf16<<<g1, 256, 0, stream>>>(featB, qkvwB, qkv_b, qkvB,
                                    BATCH * S_LEN, QKV_N, DMODEL, 1);

  dim3 g2(S_LEN / 64, BATCH * NHEADS);
  attn_flash<<<g2, 128, 0, stream>>>(qkvB, attnB, causal);

  dim3 g3(BATCH * S_LEN / 128, DMODEL / 128);
  gemm_bf16<<<g3, 256, 0, stream>>>(attnB, outwB, out_b, d_out,
                                    BATCH * S_LEN, DMODEL, DMODEL, 0);
}
